// MoE_23106924052515
// MI455X (gfx1250) — compile-verified, matrix operand order checked by reading, not run on hardware
//
#include <hip/hip_runtime.h>

#define DIMS 1024
#define HIDS 2048
#define NE   8
#define NK   2
#define LR   16
#define NT   2048   // B_*S_ tokens

typedef __bf16 bf16;
typedef __attribute__((ext_vector_type(4)))  __bf16 v4bf;
typedef __attribute__((ext_vector_type(8)))  __bf16 v8bf;
typedef __attribute__((ext_vector_type(16))) __bf16 v16bf;
typedef __attribute__((ext_vector_type(8)))  float  v8f;

// ---------------------------------------------------------------- helpers
__device__ __forceinline__ float wave_sum(float v) {
#pragma unroll
  for (int off = 16; off > 0; off >>= 1) v += __shfl_down(v, off, 32);
  return v; // valid in lane 0
}

// A/B 16-bit WMMA fragment from a row-major [rows][32] bf16 LDS tile.
// ISA 7.12.2: lane (l&15) holds row/col (l&15); lanes 0-15 hold K 0..7,16..23,
// lanes 16-31 hold K 8..15,24..31  ->  two contiguous 8-half (16B) chunks.
__device__ __forceinline__ v16bf load_frag(const bf16* rowbase, int kg) {
  v8bf lo = *(const v8bf*)(rowbase + (kg << 3));
  v8bf hi = *(const v8bf*)(rowbase + (kg << 3) + 16);
  v16bf r;
#pragma unroll
  for (int j = 0; j < 8; ++j) { r[j] = lo[j]; r[8 + j] = hi[j]; }
  return r;
}

__device__ __forceinline__ v8bf cvt8(float4 lo, float4 hi) {
  v8bf o;
  o[0] = (bf16)lo.x; o[1] = (bf16)lo.y; o[2] = (bf16)lo.z; o[3] = (bf16)lo.w;
  o[4] = (bf16)hi.x; o[5] = (bf16)hi.y; o[6] = (bf16)hi.z; o[7] = (bf16)hi.w;
  return o;
}

// ---------------------------------------------------------------- kernels
__global__ __launch_bounds__(256) void cvt_x_bf16(const float* __restrict__ x,
                                                  bf16* __restrict__ xb) {
  size_t idx = ((size_t)blockIdx.x * blockDim.x + threadIdx.x) * 4;
  float4 v = *(const float4*)(x + idx);
  v4bf o; o[0] = (bf16)v.x; o[1] = (bf16)v.y; o[2] = (bf16)v.z; o[3] = (bf16)v.w;
  *(v4bf*)(xb + idx) = o;
}

// One wave per token: scores = softmax(x @ Wg^T), top-2, compact per-expert lists.
__global__ __launch_bounds__(32) void routing_kernel(
    const float* __restrict__ x, const float* __restrict__ Wg,
    int* __restrict__ counts, int* __restrict__ tokidx, float* __restrict__ gateval) {
  const int t = blockIdx.x;
  const int lane = threadIdx.x;
  float xv[DIMS / 32];
#pragma unroll
  for (int i = 0; i < DIMS / 32; ++i) xv[i] = x[(size_t)t * DIMS + i * 32 + lane];
  float s[NE];
#pragma unroll
  for (int e = 0; e < NE; ++e) {
    float acc = 0.f;
#pragma unroll
    for (int i = 0; i < DIMS / 32; ++i) acc += xv[i] * Wg[e * DIMS + i * 32 + lane];
    acc = wave_sum(acc);
    s[e] = __shfl(acc, 0, 32);
  }
  if (lane == 0) {
    float mx = s[0];
    for (int e = 1; e < NE; ++e) mx = fmaxf(mx, s[e]);
    float p[NE];
    for (int e = 0; e < NE; ++e) p[e] = __expf(s[e] - mx);
    int e0 = 0;
    for (int e = 1; e < NE; ++e) if (p[e] > p[e0]) e0 = e;   // first-index tie break
    int e1 = (e0 == 0) ? 1 : 0;
    for (int e = 0; e < NE; ++e) if (e != e0 && p[e] > p[e1]) e1 = e;
    float w0 = p[e0], w1 = p[e1], ws = w0 + w1;  // softmax denom cancels in ratio
    w0 /= ws; w1 /= ws;
    int q0 = atomicAdd(&counts[e0], 1);
    tokidx[e0 * NT + q0] = t; gateval[e0 * NT + q0] = w0;
    int q1 = atomicAdd(&counts[e1], 1);
    tokidx[e1 * NT + q1] = t; gateval[e1 * NT + q1] = w1;
  }
}

__global__ void scan_kernel(const int* __restrict__ counts, int* __restrict__ offsets) {
  if (threadIdx.x == 0 && blockIdx.x == 0) {
    int acc = 0;
    for (int e = 0; e < NE; ++e) { offsets[e] = acc; acc += counts[e]; }
    offsets[NE] = acc;
  }
}

// u1/u3 = x @ A^T per selected (expert, token) slot. One wave per slot.
__global__ __launch_bounds__(256) void lora_u13_kernel(
    const float* __restrict__ x, const float* __restrict__ A1, const float* __restrict__ A3,
    const int* __restrict__ counts, const int* __restrict__ offsets,
    const int* __restrict__ tokidx, float* __restrict__ u1, float* __restrict__ u3) {
  const int e = blockIdx.y;
  const int slot = blockIdx.x * 8 + (threadIdx.x >> 5);
  const int lane = threadIdx.x & 31;
  if (slot >= counts[e]) return;
  const int t = tokidx[e * NT + slot];
  const int row = offsets[e] + slot;
  const float* A1e = A1 + (size_t)e * LR * DIMS;
  const float* A3e = A3 + (size_t)e * LR * DIMS;
  float a1[LR] = {}, a3[LR] = {};
  for (int i = 0; i < DIMS / 32; ++i) {
    int d = i * 32 + lane;
    float xd = x[(size_t)t * DIMS + d];
#pragma unroll
    for (int r = 0; r < LR; ++r) {
      a1[r] += xd * A1e[r * DIMS + d];
      a3[r] += xd * A3e[r * DIMS + d];
    }
  }
#pragma unroll
  for (int r = 0; r < LR; ++r) {
    float v1 = wave_sum(a1[r]);
    float v3 = wave_sum(a3[r]);
    if (lane == 0) { u1[(size_t)row * LR + r] = v1; u3[(size_t)row * LR + r] = v3; }
  }
}

// u2 = g @ A2^T (g is bf16).
__global__ __launch_bounds__(256) void lora_u2_kernel(
    const bf16* __restrict__ g, const float* __restrict__ A2,
    const int* __restrict__ counts, const int* __restrict__ offsets,
    float* __restrict__ u2) {
  const int e = blockIdx.y;
  const int slot = blockIdx.x * 8 + (threadIdx.x >> 5);
  const int lane = threadIdx.x & 31;
  if (slot >= counts[e]) return;
  const int row = offsets[e] + slot;
  const bf16* gr = g + (size_t)row * HIDS;
  const float* A2e = A2 + (size_t)e * LR * HIDS;
  float a[LR] = {};
  for (int i = 0; i < HIDS / 32; ++i) {
    int d = i * 32 + lane;
    float gv = (float)gr[d];
#pragma unroll
    for (int r = 0; r < LR; ++r) a[r] += gv * A2e[r * HIDS + d];
  }
#pragma unroll
  for (int r = 0; r < LR; ++r) {
    float v = wave_sum(a[r]);
    if (lane == 0) u2[(size_t)row * LR + r] = v;
  }
}

// g = silu(x W1^T + b1 + u1 B1^T) * (x W3^T + b3 + u3 B3^T)   [WMMA bf16]
// 256 threads = 8 waves (2x4 grid), block tile 64x128, 32x32 per wave,
// double-buffered LDS: one barrier per K-step, 8 WMMAs/wave/step.
__global__ __launch_bounds__(256) void ffn_in_kernel(
    const bf16* __restrict__ xb,
    const float* __restrict__ W1, const float* __restrict__ b1, const float* __restrict__ B1,
    const float* __restrict__ W3, const float* __restrict__ b3, const float* __restrict__ B3,
    const float* __restrict__ u1, const float* __restrict__ u3,
    const int* __restrict__ counts, const int* __restrict__ offsets,
    const int* __restrict__ tokidx, bf16* __restrict__ g) {
  const int e = blockIdx.z;
  const int cnt = counts[e];
  const int m0 = blockIdx.y * 64;
  if (m0 >= cnt) return;                       // block-uniform early exit
  const int n0 = blockIdx.x * 128;
  const int goff = offsets[e];

  __shared__ __align__(16) bf16 Alds[2][64][32];
  __shared__ __align__(16) bf16 B1lds[2][128][32];
  __shared__ __align__(16) bf16 B3lds[2][128][32];
  __shared__ int tok[64];

  const int tid = threadIdx.x;
  if (tid < 64) {
    int r = m0 + tid;
    tok[tid] = tokidx[e * NT + (r < cnt ? r : cnt - 1)];
  }
  __syncthreads();                              // tok needed by A prefetch

  const int lane = tid & 31;
  const int wid  = tid >> 5;                    // 0..7
  const int wm = wid >> 2, wn = wid & 3;        // 2x4 waves of 32x32 tiles
  const int frow = lane & 15, kg = lane >> 4;

  // cooperative fill indices
  const int aar = tid >> 2, aac = (tid & 3) * 8;   // A: 8 bf16 per thread
  const int bbr = tid >> 1, bbc = (tid & 1) * 16;  // B: 16 elems per thread

  const float* W1e = W1 + ((size_t)e * HIDS + n0) * DIMS;
  const float* W3e = W3 + ((size_t)e * HIDS + n0) * DIMS;
  const bf16*  Axp = xb + (size_t)tok[aar] * DIMS + aac;
  const float* w1p = W1e + (size_t)bbr * DIMS + bbc;
  const float* w3p = W3e + (size_t)bbr * DIMS + bbc;

  v8bf ra, rb1[2], rb3[2];                      // register prefetch stage
  auto load_stage = [&](int k0) {
    __builtin_prefetch(w1p + k0 + 64, 0, 0);    // global_prefetch_b8: next-next tile
    __builtin_prefetch(w3p + k0 + 64, 0, 0);
    ra = *(const v8bf*)(Axp + k0);
#pragma unroll
    for (int h = 0; h < 2; ++h) {
      float4 lo1 = *(const float4*)(w1p + k0 + h * 8);
      float4 hi1 = *(const float4*)(w1p + k0 + h * 8 + 4);
      rb1[h] = cvt8(lo1, hi1);
      float4 lo3 = *(const float4*)(w3p + k0 + h * 8);
      float4 hi3 = *(const float4*)(w3p + k0 + h * 8 + 4);
      rb3[h] = cvt8(lo3, hi3);
    }
  };
  auto store_stage = [&](int buf) {
    *(v8bf*)&Alds[buf][aar][aac] = ra;
    *(v8bf*)&B1lds[buf][bbr][bbc]     = rb1[0];
    *(v8bf*)&B1lds[buf][bbr][bbc + 8] = rb1[1];
    *(v8bf*)&B3lds[buf][bbr][bbc]     = rb3[0];
    *(v8bf*)&B3lds[buf][bbr][bbc + 8] = rb3[1];
  };

  v8f acc1[2][2] = {}; v8f acc3[2][2] = {};

  load_stage(0);
  store_stage(0);
  __syncthreads();

  int buf = 0;
  for (int k0 = 0; k0 < DIMS; k0 += 32) {
    const bool has_next = (k0 + 32) < DIMS;
    if (has_next) load_stage(k0 + 32);          // global prefetch into regs

    v16bf af[2], f1[2], f3[2];
#pragma unroll
    for (int s = 0; s < 2; ++s) {
      af[s] = load_frag(&Alds [buf][wm * 32 + s * 16 + frow][0], kg);
      f1[s] = load_frag(&B1lds[buf][wn * 32 + s * 16 + frow][0], kg);
      f3[s] = load_frag(&B3lds[buf][wn * 32 + s * 16 + frow][0], kg);
    }
#pragma unroll
    for (int sm = 0; sm < 2; ++sm)
#pragma unroll
      for (int sn = 0; sn < 2; ++sn) {
        acc1[sm][sn] = __builtin_amdgcn_wmma_f32_16x16x32_bf16(
            false, af[sm], false, f1[sn], (short)0, acc1[sm][sn], false, false);
        acc3[sm][sn] = __builtin_amdgcn_wmma_f32_16x16x32_bf16(
            false, af[sm], false, f3[sn], (short)0, acc3[sm][sn], false, false);
      }

    if (has_next) store_stage(buf ^ 1);         // write other buffer: no read alias
    __syncthreads();                            // single barrier per K-step
    buf ^= 1;
  }

  // epilogue: bias + rank-16 LoRA + silu gate, store g (bf16)
#pragma unroll
  for (int sn = 0; sn < 2; ++sn) {
    const int col = n0 + wn * 32 + sn * 16 + frow;
    const float bias1 = b1[e * HIDS + col];
    const float bias3 = b3[e * HIDS + col];
    float br1[LR], br3[LR];
    const float* B1r = B1 + ((size_t)e * HIDS + col) * LR;
    const float* B3r = B3 + ((size_t)e * HIDS + col) * LR;
#pragma unroll
    for (int r = 0; r < LR; ++r) { br1[r] = B1r[r]; br3[r] = B3r[r]; }
#pragma unroll
    for (int sm = 0; sm < 2; ++sm) {
#pragma unroll
      for (int i = 0; i < 8; ++i) {             // C/D layout: row = i + 8*kg
        int mrow = wm * 32 + sm * 16 + i + 8 * kg;
        int slot = m0 + mrow;
        if (slot < cnt) {
          float h1 = acc1[sm][sn][i] + bias1;
          float h3 = acc3[sm][sn][i] + bias3;
          const float* u1r = u1 + (size_t)(goff + slot) * LR;
          const float* u3r = u3 + (size_t)(goff + slot) * LR;
#pragma unroll
          for (int r = 0; r < LR; ++r) { h1 += u1r[r] * br1[r]; h3 += u3r[r] * br3[r]; }
          float sg = h1 / (1.f + __expf(-h1));  // silu
          g[(size_t)(goff + slot) * HIDS + col] = (bf16)(sg * h3);
        }
      }
    }
  }
}

// y = g W2^T + b2 + u2 B2^T ; out[t] += gate * y   [WMMA bf16 + atomic combine]
// 256 threads = 8 waves (2x4), block tile 64x128, double-buffered LDS.
__global__ __launch_bounds__(256) void ffn_out_kernel(
    const bf16* __restrict__ g,
    const float* __restrict__ W2, const float* __restrict__ b2, const float* __restrict__ B2,
    const float* __restrict__ u2,
    const int* __restrict__ counts, const int* __restrict__ offsets,
    const int* __restrict__ tokidx, const float* __restrict__ gateval,
    float* __restrict__ out) {
  const int e = blockIdx.z;
  const int cnt = counts[e];
  const int m0 = blockIdx.y * 64;
  if (m0 >= cnt) return;
  const int n0 = blockIdx.x * 128;
  const int goff = offsets[e];

  __shared__ __align__(16) bf16 Alds[2][64][32];
  __shared__ __align__(16) bf16 Blds[2][128][32];
  __shared__ int   tok[64];
  __shared__ float gat[64];

  const int tid = threadIdx.x;
  if (tid < 64) {
    int r = m0 + tid;
    int rc = r < cnt ? r : cnt - 1;
    tok[tid] = tokidx[e * NT + rc];
    gat[tid] = gateval[e * NT + rc];
  }
  const int lane = tid & 31;
  const int wid  = tid >> 5;
  const int wm = wid >> 2, wn = wid & 3;
  const int frow = lane & 15, kg = lane >> 4;
  const int aar = tid >> 2, aac = (tid & 3) * 8;
  const int bbr = tid >> 1, bbc = (tid & 1) * 16;

  const float* W2e = W2 + ((size_t)e * DIMS + n0) * HIDS;
  const int arow = goff + (m0 + aar < cnt ? m0 + aar : cnt - 1); // g rows contiguous
  const bf16*  Asrc = g + (size_t)arow * HIDS + aac;
  const float* w2p  = W2e + (size_t)bbr * HIDS + bbc;

  v8bf ra, rb[2];
  auto load_stage = [&](int k0) {
    __builtin_prefetch(w2p + k0 + 64, 0, 0);
    ra = *(const v8bf*)(Asrc + k0);
#pragma unroll
    for (int h = 0; h < 2; ++h) {
      float4 lo = *(const float4*)(w2p + k0 + h * 8);
      float4 hi = *(const float4*)(w2p + k0 + h * 8 + 4);
      rb[h] = cvt8(lo, hi);
    }
  };
  auto store_stage = [&](int buf) {
    *(v8bf*)&Alds[buf][aar][aac] = ra;
    *(v8bf*)&Blds[buf][bbr][bbc]     = rb[0];
    *(v8bf*)&Blds[buf][bbr][bbc + 8] = rb[1];
  };

  v8f acc[2][2] = {};

  load_stage(0);
  store_stage(0);
  __syncthreads();

  int buf = 0;
  for (int k0 = 0; k0 < HIDS; k0 += 32) {
    const bool has_next = (k0 + 32) < HIDS;
    if (has_next) load_stage(k0 + 32);

    v16bf af[2], bf2[2];
#pragma unroll
    for (int s = 0; s < 2; ++s) {
      af [s] = load_frag(&Alds[buf][wm * 32 + s * 16 + frow][0], kg);
      bf2[s] = load_frag(&Blds[buf][wn * 32 + s * 16 + frow][0], kg);
    }
#pragma unroll
    for (int sm = 0; sm < 2; ++sm)
#pragma unroll
      for (int sn = 0; sn < 2; ++sn)
        acc[sm][sn] = __builtin_amdgcn_wmma_f32_16x16x32_bf16(
            false, af[sm], false, bf2[sn], (short)0, acc[sm][sn], false, false);

    if (has_next) store_stage(buf ^ 1);
    __syncthreads();
    buf ^= 1;
  }

#pragma unroll
  for (int sn = 0; sn < 2; ++sn) {
    const int col = n0 + wn * 32 + sn * 16 + frow;
    const float bias2 = b2[e * DIMS + col];
    float br2[LR];
    const float* B2r = B2 + ((size_t)e * DIMS + col) * LR;
#pragma unroll
    for (int r = 0; r < LR; ++r) br2[r] = B2r[r];
#pragma unroll
    for (int sm = 0; sm < 2; ++sm) {
#pragma unroll
      for (int i = 0; i < 8; ++i) {
        int mrow = wm * 32 + sm * 16 + i + 8 * kg;
        int slot = m0 + mrow;
        if (slot < cnt) {
          float y = acc[sm][sn][i] + bias2;
          const float* u2r = u2 + (size_t)(goff + slot) * LR;
#pragma unroll
          for (int r = 0; r < LR; ++r) y += u2r[r] * br2[r];
          atomicAdd(&out[(size_t)tok[mrow] * DIMS + col], gat[mrow] * y);
        }
      }
    }
  }
}

// ---------------------------------------------------------------- launch
extern "C" void kernel_launch(void* const* d_in, const int* in_sizes, int n_in,
                              void* d_out, int out_size, void* d_ws, size_t ws_size,
                              hipStream_t stream) {
  const float* x  = (const float*)d_in[0];
  const float* Wg = (const float*)d_in[1];
  const float* W1 = (const float*)d_in[2];
  const float* b1 = (const float*)d_in[3];
  const float* A1 = (const float*)d_in[4];
  const float* B1 = (const float*)d_in[5];
  const float* W2 = (const float*)d_in[6];
  const float* b2 = (const float*)d_in[7];
  const float* A2 = (const float*)d_in[8];
  const float* B2 = (const float*)d_in[9];
  const float* W3 = (const float*)d_in[10];
  const float* b3 = (const float*)d_in[11];
  const float* A3 = (const float*)d_in[12];
  const float* B3 = (const float*)d_in[13];
  float* out = (float*)d_out;

  char* ws = (char*)d_ws;
  size_t off = 0;
  auto alloc = [&](size_t bytes) -> void* {
    void* p = ws + off;
    off = (off + bytes + 255) & ~(size_t)255;
    return p;
  };
  int*   counts  = (int*)  alloc(NE * sizeof(int));
  int*   offsets = (int*)  alloc((NE + 1) * sizeof(int));
  int*   tokidx  = (int*)  alloc((size_t)NE * NT * sizeof(int));
  float* gateval = (float*)alloc((size_t)NE * NT * sizeof(float));
  bf16*  xb      = (bf16*) alloc((size_t)NT * DIMS * sizeof(bf16));
  float* u1      = (float*)alloc((size_t)NT * NK * LR * sizeof(float));
  float* u3      = (float*)alloc((size_t)NT * NK * LR * sizeof(float));
  float* u2      = (float*)alloc((size_t)NT * NK * LR * sizeof(float));
  bf16*  g       = (bf16*) alloc((size_t)NT * NK * HIDS * sizeof(bf16));
  (void)ws_size; (void)in_sizes; (void)n_in;

  hipMemsetAsync(counts, 0, NE * sizeof(int), stream);
  hipMemsetAsync(out, 0, (size_t)out_size * sizeof(float), stream);

  cvt_x_bf16<<<(NT * DIMS) / (256 * 4), 256, 0, stream>>>(x, xb);
  routing_kernel<<<NT, 32, 0, stream>>>(x, Wg, counts, tokidx, gateval);
  scan_kernel<<<1, 32, 0, stream>>>(counts, offsets);
  lora_u13_kernel<<<dim3(NT / 8, NE), 256, 0, stream>>>(x, A1, A3, counts, offsets,
                                                        tokidx, u1, u3);
  ffn_in_kernel<<<dim3(HIDS / 128, NT / 64, NE), 256, 0, stream>>>(
      xb, W1, b1, B1, W3, b3, B3, u1, u3, counts, offsets, tokidx, g);
  lora_u2_kernel<<<dim3(NT / 8, NE), 256, 0, stream>>>(g, A2, counts, offsets, u2);
  ffn_out_kernel<<<dim3(DIMS / 128, NT / 64, NE), 256, 0, stream>>>(
      g, W2, b2, B2, u2, counts, offsets, tokidx, gateval, out);
}
